// MemoryGraph_14293651161881
// MI455X (gfx1250) — matrix operand order repaired and checked by reading, not compiled, not run on hardware
//
#include <hip/hip_runtime.h>
#include <hip/hip_bf16.h>
#include <math.h>

#define B_  16
#define T_  32
#define NC_ 16
#define N_  128
#define D_  64
#define A_  4
#define HS_ 256

typedef __attribute__((ext_vector_type(16))) __bf16          v16bf;
typedef __attribute__((ext_vector_type(16))) unsigned short  v16us;
typedef __attribute__((ext_vector_type(8)))  float           v8f;

// ---------- scalar helpers ---------------------------------------------------

__device__ __forceinline__ unsigned short f2bf(float f) {
  unsigned int u = __builtin_bit_cast(unsigned int, f);
  u += 0x7FFFu + ((u >> 16) & 1u);            // round-to-nearest-even
  return (unsigned short)(u >> 16);
}

// CDNA5 has a hardware transcendental TANH; use it if the toolchain exposes it.
#if __has_builtin(__builtin_amdgcn_tanhf)
#define TANHF(x) __builtin_amdgcn_tanhf(x)
#elif __has_builtin(__builtin_amdgcn_tanh_f32)
#define TANHF(x) __builtin_amdgcn_tanh_f32(x)
#else
#define TANHF(x) tanhf(x)
#endif

__device__ __forceinline__ v8f splat8(float x) {
  v8f c;
#pragma unroll
  for (int r = 0; r < 8; ++r) c[r] = x;
  return c;
}

__device__ __forceinline__ v8f wmma_bf16(v16bf a, v16bf b, v8f c) {
  return __builtin_amdgcn_wmma_f32_16x16x32_bf16(
      false, a, false, b, (short)0, c, false, false);
}

// ---------- async global -> LDS (CDNA5, ASYNCcnt-tracked) --------------------

__device__ __forceinline__ void async_ld_b128(unsigned lds_off, const void* gptr) {
  unsigned long long ga = (unsigned long long)gptr;
  asm volatile("global_load_async_to_lds_b128 %0, %1, off"
               :: "v"(lds_off), "v"(ga) : "memory");
}

__device__ __forceinline__ void wait_async0() {
#if __has_builtin(__builtin_amdgcn_s_wait_asynccnt)
  __builtin_amdgcn_s_wait_asynccnt(0);
#else
  asm volatile("s_wait_asynccnt 0" ::: "memory");
#endif
}

// copy 96KB (49152 bf16) global -> LDS, 16B per lane-op, 24 ops/thread
__device__ __forceinline__ void stage96(unsigned short* dstLds,
                                        const unsigned short* src, int tid) {
  unsigned base = (unsigned)(unsigned long long)(void*)dstLds;
  const char* s = (const char*)src;
#pragma unroll
  for (int i = 0; i < 24; ++i) {
    const int off = (tid + i * 256) * 16;
    async_ld_b128(base + (unsigned)off, s + off);
  }
}

// ---------- WMMA fragment loaders (layouts per CDNA5 ISA 7.12.2) -------------

// A fragment (16x32 MxK), row-major f32 source, base pre-offset to [m0][k0]
__device__ __forceinline__ v16bf load_a_f32(const float* p0, int ld, int lane) {
  const float* p = p0 + (lane & 15) * ld + ((lane >> 4) << 3);
  v16us r;
#pragma unroll
  for (int v = 0; v < 4; ++v) {
    r[2*v]       = f2bf(p[2*v]);
    r[2*v + 1]   = f2bf(p[2*v + 1]);
    r[8 + 2*v]   = f2bf(p[16 + 2*v]);
    r[8 + 2*v+1] = f2bf(p[16 + 2*v + 1]);
  }
  return __builtin_bit_cast(v16bf, r);
}

__device__ __forceinline__ v16bf load_a_bf16(const unsigned short* p0, int ld, int lane) {
  const unsigned short* p = p0 + (lane & 15) * ld + ((lane >> 4) << 3);
  v16us r;
#pragma unroll
  for (int v = 0; v < 4; ++v) {
    r[2*v]       = p[2*v];
    r[2*v + 1]   = p[2*v + 1];
    r[8 + 2*v]   = p[16 + 2*v];
    r[8 + 2*v+1] = p[16 + 2*v + 1];
  }
  return __builtin_bit_cast(v16bf, r);
}

// B fragment (32x16 KxN), storage N-major rows: elem(k,n)=base[n*ld+k] (contiguous)
__device__ __forceinline__ v16bf load_b_nk(const unsigned short* p0, int ld, int lane) {
  const unsigned short* p = p0 + (lane & 15) * ld + ((lane >> 4) << 3);
  v16us r;
#pragma unroll
  for (int i = 0; i < 16; ++i) r[i] = p[(i & 7) + ((i >> 3) << 4)];
  return __builtin_bit_cast(v16bf, r);
}

// C tile store (f32 16x16): lane l holds (M = 8*(l>>4)+r, N = l&15)
__device__ __forceinline__ void store_c_bf16(unsigned short* p0, int ld, int lane, v8f c) {
  unsigned short* p = p0 + (((lane >> 4) << 3) * ld) + (lane & 15);
#pragma unroll
  for (int r = 0; r < 8; ++r) p[r * ld] = f2bf(c[r]);
}

__device__ __forceinline__ void store_c_tanh_bf16(unsigned short* p0, int ld, int lane, v8f c) {
  unsigned short* p = p0 + (((lane >> 4) << 3) * ld) + (lane & 15);
#pragma unroll
  for (int r = 0; r < 8; ++r) p[r * ld] = f2bf(TANHF(c[r]));
}

// ---------- weight bf16 staging area in d_ws ---------------------------------
// ushort layout: [ w1r | w1h | w2 ][ mw1h | mw1id | mw2 ][ nid(NC x 128 x 64) ]
#define OFF_W1R   0
#define OFF_W1H   16384
#define OFF_W2    32768
#define OFF_MW1H  49152
#define OFF_MW1I  65536
#define OFF_MW2   81920
#define OFF_NID   98304
#define WS_TOTAL  229376

__global__ __launch_bounds__(256) void convert_weights(
    const float* __restrict__ sw1, const float* __restrict__ sw2,
    const float* __restrict__ mw1, const float* __restrict__ mw2,
    const float* __restrict__ nid, unsigned short* __restrict__ ws)
{
  int i = blockIdx.x * 256 + threadIdx.x;
  if (i >= WS_TOTAL) return;
  float v;
  if (i < OFF_W1H)       { int j = i;            v = sw1[(j >> 6) * 128 + (j & 63)]; }
  else if (i < OFF_W2)   { int j = i - OFF_W1H;  v = sw1[(j >> 6) * 128 + 64 + (j & 63)]; }
  else if (i < OFF_MW1H) { v = sw2[i - OFF_W2]; }
  else if (i < OFF_MW1I) { int j = i - OFF_MW1H; v = mw1[(j >> 6) * 128 + (j & 63)]; }
  else if (i < OFF_MW2)  { int j = i - OFF_MW1I; v = mw1[(j >> 6) * 128 + 64 + (j & 63)]; }
  else if (i < OFF_NID)  { v = mw2[i - OFF_MW2]; }
  else                   { v = nid[i - OFF_NID]; }
  ws[i] = f2bf(v);
}

// ---------- persistent per-cell recurrent kernel -----------------------------
// One block per (b,c) cell, 256 threads = 8 wave32; all T=32 steps in-kernel.
// LDS-resident state: Wh=W+heb (f32), h (f32), msg (+transpose) bf16.
// Per step the two 96KB weight sets are async-staged into one shared buffer.

#define LDS_BYTES 258304

__global__ __launch_bounds__(256) void memgraph_kernel(
    const float* __restrict__ gX,     const float* __restrict__ gH0,
    const float* __restrict__ gMsg0,  const float* __restrict__ gW,
    const float* __restrict__ gHeb0,  const float* __restrict__ gDecay,
    const float* __restrict__ gB1,    const float* __restrict__ gB2,
    const float* __restrict__ gMB1,   const float* __restrict__ gMB2,
    const float* __restrict__ gInjW,  const float* __restrict__ gInjB,
    const float* __restrict__ gLogit, const unsigned short* __restrict__ wsBf,
    float* __restrict__ gOut)
{
  extern __shared__ __align__(16) char smem_raw[];
  float*          sWh   = (float*)smem_raw;                  // 128x128 f32
  float*          sH    = sWh + N_ * N_;                     // 128x64  f32
  unsigned short* sWgt  = (unsigned short*)(sH + N_ * D_);   // 3x(16384) bf16 staged
  unsigned short* sMsg  = sWgt + 49152;                      // 128x64 bf16
  unsigned short* sMsgT = sMsg + N_ * D_;                    // 64x128 bf16 (transpose)
  unsigned short* sRecv = sMsgT + N_ * D_;                   // 128x64 bf16
  unsigned short* sStage= sRecv + N_ * D_;                   // 8 x (16x32) bf16
  float*          sXt   = (float*)(sStage + 8 * 512);        // 64
  float*          sDecay= sXt + 64;                          // 128
  float*          sB1   = sDecay + 128;                      // 256
  float*          sB2   = sB1 + 256;                         // 64
  float*          sMB1  = sB2 + 64;                          // 256
  float*          sMB2  = sMB1 + 256;                        // 64
  float*          sInjB = sMB2 + 64;                         // 256

  const int tid  = threadIdx.x;
  const int lane = tid & 31;          // wave32
  const int wv   = tid >> 5;          // 8 waves
  const int m0   = wv * 16;
  const int cell = blockIdx.x;
  const int b    = cell / NC_;
  const int cc   = cell % NC_;

  const unsigned short* wsA  = wsBf + OFF_W1R;    // w1r|w1h|w2   (96KB)
  const unsigned short* wsB  = wsBf + OFF_MW1H;   // mw1h|mw1id|mw2 (96KB)
  const unsigned short* gNID = wsBf + OFF_NID + cc * (N_ * D_);

  const float* Wcell = gW    + (size_t)cell * (N_ * N_);
  const float* Hcell = gHeb0 + (size_t)cell * (N_ * N_);
  const float* h0c   = gH0   + (size_t)cell * (N_ * D_);
  const float* mg0c  = gMsg0 + (size_t)cell * (N_ * D_);

  // ---- prologue: stage state + constants
  for (int i = tid; i < N_ * N_; i += 256) sWh[i] = Wcell[i] + Hcell[i];
  for (int i = tid; i < N_ * D_; i += 256) {
    const float mv = mg0c[i];
    const unsigned short mb = f2bf(mv);
    sH[i] = h0c[i];
    sMsg[i] = mb;
    sMsgT[(i & 63) * N_ + (i >> 6)] = mb;
  }
  if (tid < 128) sDecay[tid] = gDecay[cell * N_ + tid];
  sB1[tid]  = gB1[tid];
  sMB1[tid] = gMB1[tid];
  if (tid < 64) { sB2[tid] = gB2[tid]; sMB2[tid] = gMB2[tid]; }
  sInjB[tid] = gInjB[cc * 256 + tid];
  const float gamma = 1.0f / (1.0f + expf(-gLogit[cc]));
  __syncthreads();

  unsigned short* stg = sStage + wv * 512;
  const int hl = lane >> 4, nl = lane & 15;

  for (int t = 0; t < T_; ++t) {
    // kick off async staging of weight set A; overlaps injection + matmul1
    stage96(sWgt, wsA, tid);

    if (tid < 64) sXt[tid] = gX[(b * T_ + t) * D_ + tid];
    __syncthreads();
    { // injection into first A rows of h (h_aug, in place)
      const float* wrow = gInjW + ((size_t)cc * 256 + tid) * D_;
      float s = sInjB[tid];
#pragma unroll 8
      for (int d = 0; d < D_; ++d) s += wrow[d] * sXt[d];
      sH[(tid >> 6) * D_ + (tid & 63)] += TANHF(s);
    }
    __syncthreads();                                   // (a)

    // ---- matmul1: recv = Wh @ msg (K=128), all-LDS, overlaps weight stage
    {
      v16bf aW0 = load_a_f32(sWh + m0 * N_ +  0, N_, lane);
      v16bf aW1 = load_a_f32(sWh + m0 * N_ + 32, N_, lane);
      v16bf aW2 = load_a_f32(sWh + m0 * N_ + 64, N_, lane);
      v16bf aW3 = load_a_f32(sWh + m0 * N_ + 96, N_, lane);
#pragma unroll
      for (int n = 0; n < 4; ++n) {
        v8f c = splat8(0.0f);
        c = wmma_bf16(aW0, load_b_nk(sMsgT + (n * 16) * N_ +  0, N_, lane), c);
        c = wmma_bf16(aW1, load_b_nk(sMsgT + (n * 16) * N_ + 32, N_, lane), c);
        c = wmma_bf16(aW2, load_b_nk(sMsgT + (n * 16) * N_ + 64, N_, lane), c);
        c = wmma_bf16(aW3, load_b_nk(sMsgT + (n * 16) * N_ + 96, N_, lane), c);
        store_c_bf16(sRecv + m0 * D_ + n * 16, D_, lane, c);
      }
    }
    wait_async0();
    __syncthreads();                                   // (b) weights A ready

    // ---- fused matmul2(+tanh)+matmul3: delta = tanh(recv@w1r^T+h@w1h^T+b1)@w2^T
    v8f dAcc[4];
    {
      v16bf aR0 = load_a_bf16(sRecv + m0 * D_ +  0, D_, lane);
      v16bf aR1 = load_a_bf16(sRecv + m0 * D_ + 32, D_, lane);
      v16bf aH0 = load_a_f32 (sH    + m0 * D_ +  0, D_, lane);
      v16bf aH1 = load_a_f32 (sH    + m0 * D_ + 32, D_, lane);
#pragma unroll
      for (int n = 0; n < 4; ++n) dAcc[n] = splat8(0.0f);
      for (int hp = 0; hp < 8; ++hp) {
#pragma unroll
        for (int sub = 0; sub < 2; ++sub) {
          const int ht = hp * 2 + sub;
          v8f c = splat8(sB1[ht * 16 + nl]);
          c = wmma_bf16(aR0, load_b_nk(sWgt +         (ht * 16) * D_ +  0, D_, lane), c);
          c = wmma_bf16(aH0, load_b_nk(sWgt + 16384 + (ht * 16) * D_ +  0, D_, lane), c);
          c = wmma_bf16(aR1, load_b_nk(sWgt +         (ht * 16) * D_ + 32, D_, lane), c);
          c = wmma_bf16(aH1, load_b_nk(sWgt + 16384 + (ht * 16) * D_ + 32, D_, lane), c);
          store_c_tanh_bf16(stg + sub * 16, 32, lane, c);
        }
        v16bf ah = load_a_bf16(stg, 32, lane);
#pragma unroll
        for (int n = 0; n < 4; ++n)
          dAcc[n] = wmma_bf16(ah, load_b_nk(sWgt + 32768 + (n * 16) * HS_ + hp * 32, HS_, lane), dAcc[n]);
      }
    }
    // ---- gated h update + readout rows A..2A-1
    {
#pragma unroll
      for (int n = 0; n < 4; ++n) {
        const int col = n * 16 + nl;
#pragma unroll
        for (int r = 0; r < 8; ++r) {
          const int row = m0 + hl * 8 + r;
          const float g = sDecay[row];
          const float hn = g * sH[row * D_ + col]
                         + (1.0f - g) * TANHF(dAcc[n][r] + sB2[col]);
          sH[row * D_ + col] = hn;
          if ((unsigned)(row - A_) < (unsigned)A_)
            gOut[((size_t)(b * T_ + t)) * 4096 + cc * 256 + (row - A_) * D_ + col] = hn;
        }
      }
    }
    __syncthreads();                                   // (b2) set A fully consumed
    stage96(sWgt, wsB, tid);
    wait_async0();
    __syncthreads();                                   // (b3) weights B ready

    // ---- fused matmul4(+tanh)+matmul5: msg = tanh(tanh(h@mw1h^T+nid@mw1id^T+b1m)@mw2^T+b2m)
    v8f mAcc[4];
    {
      v16bf aN0 = load_a_f32 (sH   + m0 * D_ +  0, D_, lane);
      v16bf aN1 = load_a_f32 (sH   + m0 * D_ + 32, D_, lane);
      v16bf aI0 = load_a_bf16(gNID + m0 * D_ +  0, D_, lane);
      v16bf aI1 = load_a_bf16(gNID + m0 * D_ + 32, D_, lane);
#pragma unroll
      for (int n = 0; n < 4; ++n) mAcc[n] = splat8(0.0f);
      for (int hp = 0; hp < 8; ++hp) {
#pragma unroll
        for (int sub = 0; sub < 2; ++sub) {
          const int ht = hp * 2 + sub;
          v8f c = splat8(sMB1[ht * 16 + nl]);
          c = wmma_bf16(aN0, load_b_nk(sWgt +         (ht * 16) * D_ +  0, D_, lane), c);
          c = wmma_bf16(aI0, load_b_nk(sWgt + 16384 + (ht * 16) * D_ +  0, D_, lane), c);
          c = wmma_bf16(aN1, load_b_nk(sWgt +         (ht * 16) * D_ + 32, D_, lane), c);
          c = wmma_bf16(aI1, load_b_nk(sWgt + 16384 + (ht * 16) * D_ + 32, D_, lane), c);
          store_c_tanh_bf16(stg + sub * 16, 32, lane, c);
        }
        v16bf ah = load_a_bf16(stg, 32, lane);
#pragma unroll
        for (int n = 0; n < 4; ++n)
          mAcc[n] = wmma_bf16(ah, load_b_nk(sWgt + 32768 + (n * 16) * HS_ + hp * 32, HS_, lane), mAcc[n]);
      }
      // msg activation; write msg and its transpose
#pragma unroll
      for (int n = 0; n < 4; ++n) {
        const int col = n * 16 + nl;
#pragma unroll
        for (int r = 0; r < 8; ++r) {
          const int row = m0 + hl * 8 + r;
          const unsigned short mb = f2bf(TANHF(mAcc[n][r] + sMB2[col]));
          sMsg[row * D_ + col] = mb;
          sMsgT[col * N_ + row] = mb;
        }
      }
    }
    __syncthreads();                                   // (c) msg_new complete

    // ---- outer product + hebbian fold: Wh = g*Wh + (1-g)*(W + msg msg^T / D)
    {
      v16bf aM0 = load_a_bf16(sMsg + m0 * D_ +  0, D_, lane);
      v16bf aM1 = load_a_bf16(sMsg + m0 * D_ + 32, D_, lane);
#pragma unroll
      for (int n = 0; n < 8; ++n) {
        v8f c = splat8(0.0f);
        c = wmma_bf16(aM0, load_b_nk(sMsg + (n * 16) * D_ +  0, D_, lane), c);
        c = wmma_bf16(aM1, load_b_nk(sMsg + (n * 16) * D_ + 32, D_, lane), c);
        const int col = n * 16 + nl;
#pragma unroll
        for (int r = 0; r < 8; ++r) {
          const int row = m0 + hl * 8 + r;
          const int idx = row * N_ + col;
          sWh[idx] = gamma * sWh[idx]
                   + (1.0f - gamma) * (Wcell[idx] + c[r] * (1.0f / 64.0f));
        }
      }
    }
    __syncthreads();                                   // (d) end of step
  }
}

// ---------- launch -----------------------------------------------------------

extern "C" void kernel_launch(void* const* d_in, const int* in_sizes, int n_in,
                              void* d_out, int out_size, void* d_ws, size_t ws_size,
                              hipStream_t stream) {
  const float* x     = (const float*)d_in[0];
  const float* h0    = (const float*)d_in[1];
  const float* msg0  = (const float*)d_in[2];
  const float* W     = (const float*)d_in[3];
  const float* heb0  = (const float*)d_in[4];
  const float* decay = (const float*)d_in[5];
  const float* nid   = (const float*)d_in[6];
  const float* sw1   = (const float*)d_in[7];
  const float* sb1   = (const float*)d_in[8];
  const float* sw2   = (const float*)d_in[9];
  const float* sb2   = (const float*)d_in[10];
  const float* mw1   = (const float*)d_in[11];
  const float* mb1   = (const float*)d_in[12];
  const float* mw2   = (const float*)d_in[13];
  const float* mb2   = (const float*)d_in[14];
  const float* injw  = (const float*)d_in[15];
  const float* injb  = (const float*)d_in[16];
  const float* logit = (const float*)d_in[17];
  unsigned short* ws = (unsigned short*)d_ws;
  float* out         = (float*)d_out;

  convert_weights<<<(WS_TOTAL + 255) / 256, 256, 0, stream>>>(sw1, sw2, mw1, mw2, nid, ws);

  (void)hipFuncSetAttribute((const void*)memgraph_kernel,
                            hipFuncAttributeMaxDynamicSharedMemorySize, LDS_BYTES);
  memgraph_kernel<<<B_ * NC_, 256, LDS_BYTES, stream>>>(
      x, h0, msg0, W, heb0, decay, sb1, sb2, mb1, mb2,
      injw, injb, logit, ws, out);
  (void)in_sizes; (void)n_in; (void)out_size; (void)ws_size;
}
